// MLA_32366873542851
// MI455X (gfx1250) — matrix operand order repaired and checked by reading, not compiled, not run on hardware
//
#include <hip/hip_runtime.h>
#include <math.h>
#include <stdint.h>

typedef __attribute__((ext_vector_type(16))) __bf16 v16bf;
typedef __attribute__((ext_vector_type(8)))  __bf16 v8bf;
typedef __attribute__((ext_vector_type(8)))  float  v8f;
typedef __attribute__((ext_vector_type(4)))  int    v4i;
typedef __attribute__((ext_vector_type(8)))  int    v8i;
typedef __attribute__((ext_vector_type(4)))  unsigned int v4u;

#define BM 128
#define BN 128
#define BK 32
#define KP 40   // padded LDS row stride (elements): 64B row + 16B pad

#if defined(__HIP_DEVICE_COMPILE__) && __has_builtin(__builtin_amdgcn_global_load_async_to_lds_b128)
#define ASYNC_COPY 1
#else
#define ASYNC_COPY 0
#endif

#if ASYNC_COPY && __has_builtin(__builtin_amdgcn_tensor_load_to_lds) && __has_builtin(__builtin_amdgcn_s_wait_tensorcnt)
#define TDM_COPY 1
#else
#define TDM_COPY 0
#endif

// amdgpu-toolchain (therock headers) exposes the 6-arg tensor builtin and is
// the only lane shipping this header; ROCm 7.2 clang-22 uses the 5-arg form.
#if __has_include(<hip/amd_detail/amd_gfx1250_TDM.h>)
#define TDM_ARITY6 1
#else
#define TDM_ARITY6 0
#endif

typedef __attribute__((address_space(1))) v4i* gbl_v4i;  // global int4*
typedef __attribute__((address_space(3))) v4i* lds_v4i;  // LDS int4*

__device__ __forceinline__ gbl_v4i to_gbl(const void* p){
  return (gbl_v4i)(uintptr_t)p;
}
__device__ __forceinline__ lds_v4i to_lds(const void* p){
  // generic LDS pointer carries the LDS byte offset in bits [31:0]
  return (lds_v4i)(unsigned)(uintptr_t)p;
}

__device__ __forceinline__ void wait_async0(){
#if ASYNC_COPY
#if __has_builtin(__builtin_amdgcn_s_wait_asynccnt)
  __builtin_amdgcn_s_wait_asynccnt(0);
#else
  asm volatile("s_wait_asynccnt 0" ::: "memory");
#endif
#endif
}

__device__ __forceinline__ v16bf cat8(v8bf lo, v8bf hi){
  v16bf r;
#pragma unroll
  for (int i=0;i<8;++i){ r[i]=lo[i]; r[i+8]=hi[i]; }
  return r;
}

// -------------------- generic batched bf16 WMMA GEMM --------------------
// C[z] = scale * A[z] * B[z]^T + bias ; z offsets split as (z>>3)*Z1 + (z&7)*Z2.
// OMODE 0: fp32 store ; 1: bf16 store ; 2: both ; 3: head-transposed V store
//   (vt[((b*8+h)*640+d)*2048 + s], bf16 only)
// Data movement: A tile via per-lane async global->LDS engine; B tile via one
// TDM tensor_load_to_lds 2D descriptor per workgroup (if toolchain exposes it).
template<int OMODE>
__global__ __launch_bounds__(256) void gemm_bf16_wmma(
    const __bf16* __restrict__ A, long lda, long aZ1, long aZ2,
    const __bf16* __restrict__ Bt, long ldb, long bZ1, long bZ2,
    const float* __restrict__ bias, float scale,
    float* __restrict__ Cf, __bf16* __restrict__ Cb,
    long ldc, long cZ1, long cZ2, int K)
{
  __shared__ __bf16 As[2][BM*KP];
  __shared__ __bf16 Bs[2][BN*KP];

  const int z = blockIdx.z;
  A  += (long)(z>>3)*aZ1 + (long)(z&7)*aZ2;
  Bt += (long)(z>>3)*bZ1 + (long)(z&7)*bZ2;
  const long coff = (long)(z>>3)*cZ1 + (long)(z&7)*cZ2;

  const int m0 = blockIdx.y * BM;
  const int n0 = blockIdx.x * BN;
  const int tid  = threadIdx.x;
  const int lane = tid & 31;
  const int w    = tid >> 5;
  const int wm   = w >> 1;      // 0..3 : 32 rows each
  const int wn   = w & 1;       // 0..1 : 64 cols each
  const int half = lane >> 4;   // K-half select per ISA lane layout
  const int l16  = lane & 15;

  v8f acc[2][4] = {};

  const int ldRow = tid >> 2;          // 0..63
  const int ldCol = (tid & 3) * 8;     // 0,8,16,24 (bf16 elems, 16B chunks)

  const __bf16* Arow0 = A  + (long)(m0 + ldRow)      * lda + ldCol;
  const __bf16* Arow1 = A  + (long)(m0 + ldRow + 64) * lda + ldCol;
  const __bf16* Brow0 = Bt + (long)(n0 + ldRow)      * ldb + ldCol;
  const __bf16* Brow1 = Bt + (long)(n0 + ldRow + 64) * ldb + ldCol;
  (void)Brow0; (void)Brow1;

  auto compute = [&](int b){
    v16bf af[2], bfr[4];
#pragma unroll
    for (int mi=0; mi<2; ++mi){
      // A frag: lane<16 -> K {0..7,16..23}; lane>=16 -> K {8..15,24..31}
      const __bf16* p = &As[b][(wm*32 + mi*16 + l16)*KP + half*8];
      af[mi] = cat8(*reinterpret_cast<const v8bf*>(p),
                    *reinterpret_cast<const v8bf*>(p+16));
    }
#pragma unroll
    for (int ni=0; ni<4; ++ni){
      // B frag: lane col = l16, lane<16 -> K {0..7,16..23}; lane>=16 -> K {8..15,24..31} (B^T rows)
      const __bf16* p = &Bs[b][(wn*64 + ni*16 + l16)*KP + half*8];
      bfr[ni] = cat8(*reinterpret_cast<const v8bf*>(p),
                     *reinterpret_cast<const v8bf*>(p+16));
    }
#pragma unroll
    for (int mi=0; mi<2; ++mi)
#pragma unroll
      for (int ni=0; ni<4; ++ni)
        acc[mi][ni] = __builtin_amdgcn_wmma_f32_16x16x32_bf16(
            false, af[mi], false, bfr[ni], (short)0, acc[mi][ni], false, false);
  };

  int buf = 0;
#if ASYNC_COPY
  auto tile_A = [&](int k0, int b){
    __builtin_amdgcn_global_load_async_to_lds_b128(
        to_gbl(Arow0 + k0), to_lds(&As[b][ldRow*KP + ldCol]), 0, 0);
    __builtin_amdgcn_global_load_async_to_lds_b128(
        to_gbl(Arow1 + k0), to_lds(&As[b][(ldRow+64)*KP + ldCol]), 0, 0);
  };
#if TDM_COPY
  // One TDM 2D-tile DMA per workgroup: 128 rows x 64B of B^T, LDS-padded
  // +16B per row (pad_interval=16 DWORDs, pad_amount=4 DWORDs -> KP=40 elems).
  auto tile_B = [&](int k0, int b){
    if (w == 0) {
      const unsigned long long ga =
          (unsigned long long)(uintptr_t)(Bt + (long)n0*ldb + k0);
      const unsigned ldsb = (unsigned)(uintptr_t)(&Bs[b][0]);
      v4u g0; v8i g1;
      g0[0] = 1u;                                  // count=1 (valid user D#)
      g0[1] = ldsb;                                // lds_addr
      g0[2] = (unsigned)(ga & 0xFFFFFFFFull);      // global_addr[31:0]
      g0[3] = (unsigned)((ga >> 32) & 0x1FFFFFFull) | (2u << 30); // addr[56:32] | type=2
      g1[0] = (1<<16) | (1<<20) | (3<<22) | (3<<25); // data_size=2B | pad_en | int=16DW | amt=4DW
      g1[1] = 0;                                   // no atomic barrier; tensor_dim0 lo16 (2^30)=0
      g1[2] = 0x4000;                              // tensor_dim0 hi16 ; tensor_dim1 lo16=0
      g1[3] = (32 << 16) | 0x4000;                 // tensor_dim1 hi16 ; tile_dim0=32 elems
      g1[4] = 128;                                 // tile_dim1=128 rows ; tile_dim2=0 (2D)
      g1[5] = (int)ldb;                            // tensor_dim0_stride (elements)
      g1[6] = 0;                                   // stride hi16 ; dim1_stride lo16
      g1[7] = 0;
      const v4i g2 = {0,0,0,0};                    // dims 2-3 unused (tile_dim2=0)
      const v4i g3 = {0,0,0,0};                    // dim 4 unused
#if TDM_ARITY6
      const v8i g4 = {0,0,0,0,0,0,0,0};
      __builtin_amdgcn_tensor_load_to_lds(g0, g1, g2, g3, g4, 0);
#else
      __builtin_amdgcn_tensor_load_to_lds(g0, g1, g2, g3, 0);
#endif
    }
  };
#else
  auto tile_B = [&](int k0, int b){
    __builtin_amdgcn_global_load_async_to_lds_b128(
        to_gbl(Brow0 + k0), to_lds(&Bs[b][ldRow*KP + ldCol]), 0, 0);
    __builtin_amdgcn_global_load_async_to_lds_b128(
        to_gbl(Brow1 + k0), to_lds(&Bs[b][(ldRow+64)*KP + ldCol]), 0, 0);
  };
#endif
  auto sync_tiles = [&](){
    wait_async0();
#if TDM_COPY
    if (w == 0) __builtin_amdgcn_s_wait_tensorcnt(0);
#endif
    __syncthreads();
  };

  tile_A(0, 0);
  tile_B(0, 0);
  sync_tiles();
  for (int k0 = 0; k0 < K; k0 += BK){
    if (k0 + BK < K){ tile_A(k0 + BK, buf ^ 1); tile_B(k0 + BK, buf ^ 1); }
    compute(buf);        // WMMA from current buffer while DMAs fill the other
    sync_tiles();
    buf ^= 1;
  }
#else
  // software-pipelined sync fallback: global->regs early, ds_store after compute
  uint4 ra0, ra1, rb0, rb1;
  auto load_regs = [&](int k0){
    ra0 = *reinterpret_cast<const uint4*>(Arow0 + k0);
    ra1 = *reinterpret_cast<const uint4*>(Arow1 + k0);
    rb0 = *reinterpret_cast<const uint4*>(Brow0 + k0);
    rb1 = *reinterpret_cast<const uint4*>(Brow1 + k0);
  };
  auto store_lds = [&](int b){
    *reinterpret_cast<uint4*>(&As[b][ldRow*KP + ldCol])      = ra0;
    *reinterpret_cast<uint4*>(&As[b][(ldRow+64)*KP + ldCol]) = ra1;
    *reinterpret_cast<uint4*>(&Bs[b][ldRow*KP + ldCol])      = rb0;
    *reinterpret_cast<uint4*>(&Bs[b][(ldRow+64)*KP + ldCol]) = rb1;
  };
  load_regs(0);
  store_lds(0);
  __syncthreads();
  for (int k0 = 0; k0 < K; k0 += BK){
    const bool more = (k0 + BK < K);
    if (more) load_regs(k0 + BK);
    compute(buf);
    if (more) store_lds(buf ^ 1);
    __syncthreads();
    buf ^= 1;
  }
#endif

  // epilogue: C layout lane l16 -> N, half*8 + r -> M
#pragma unroll
  for (int mi=0; mi<2; ++mi){
#pragma unroll
    for (int ni=0; ni<4; ++ni){
      const int col  = n0 + wn*64 + ni*16 + l16;
      const int row0 = m0 + wm*32 + mi*16 + half*8;
      const float bv = bias ? bias[col] : 0.0f;
#pragma unroll
      for (int r=0; r<8; ++r){
        const float v = acc[mi][ni][r] * scale + bv;
        const long m = row0 + r;
        if (OMODE == 3){
          const long b = m >> 11, s = m & 2047;
          const long hh = col / 640, d = col % 640;
          Cb[(((b*8 + hh)*640 + d) << 11) + s] = (__bf16)v;
        } else {
          if (OMODE == 0 || OMODE == 2) Cf[coff + m*ldc + col] = v;
          if (OMODE == 1 || OMODE == 2) Cb[coff + m*ldc + col] = (__bf16)v;
        }
      }
    }
  }
}

// -------------------- fp32 -> bf16 convert --------------------
__global__ __launch_bounds__(256) void cvt_f32_to_bf16(const float* __restrict__ x,
                                                       __bf16* __restrict__ y, long n){
  const long i = ((long)blockIdx.x*256 + threadIdx.x) * 4;
  if (i + 3 < n) {
    const float4 v = *reinterpret_cast<const float4*>(x + i);
    y[i+0]=(__bf16)v.x; y[i+1]=(__bf16)v.y; y[i+2]=(__bf16)v.z; y[i+3]=(__bf16)v.w;
  } else {
    for (long k=i; k<n; ++k) y[k]=(__bf16)x[k];
  }
}

// -------------------- fp32 W[K][N] -> bf16 W^T[N][K] --------------------
__global__ __launch_bounds__(256) void transpose_cvt(const float* __restrict__ W,
                                                     __bf16* __restrict__ Wt,
                                                     int Kd, int Nd){
  __shared__ float tile[32][33];
  const int bx = blockIdx.x * 32; // N
  const int by = blockIdx.y * 32; // K
  const int tx = threadIdx.x & 31, ty = threadIdx.x >> 5;
  for (int r = ty; r < 32; r += 8){
    const int k = by + r, n = bx + tx;
    tile[r][tx] = (k < Kd && n < Nd) ? W[(long)k*Nd + n] : 0.0f;
  }
  __syncthreads();
  for (int r = ty; r < 32; r += 8){
    const int n = bx + r, k = by + tx;
    if (n < Nd && k < Kd) Wt[(long)n*Kd + k] = (__bf16)tile[tx][r];
  }
}

// -------------------- rotary helper --------------------
// cos/sin have width 8 (D_ROPE/2): only dims 0..7 rotate, 8..15 pass through.
// inv_freq[jj] = 10000^(-0.25*jj), t = s/40
__device__ __forceinline__ float rope_rotate(const float* x16, int j, int s){
  if (j >= 8) return x16[j];
  const int jj = j & 3;
  const float ang = ((float)s * 0.025f) * __expf(-2.302585093f * (float)jj);
  const float c  = __cosf(ang), sn = __sinf(ang);
  const float partner = (j < 4) ? -x16[j+4] : x16[j-4];
  return x16[j]*c + partner*sn;
}

// q[t, h*640+d] = d<624 ? q_base[t, h*624+d] : rope(q_rot[t, h*16 + (d-624)])
__global__ __launch_bounds__(256) void assemble_q_kernel(
    const __bf16* __restrict__ qb, const float* __restrict__ qr,
    __bf16* __restrict__ q)
{
  const long i = (long)blockIdx.x * 256 + threadIdx.x;
  const long t = i / 5120;
  const int  c = (int)(i - t*5120);
  const int  h = c / 640, d = c - h*640;
  __bf16 val;
  if (d < 624) {
    val = qb[t*4992 + h*624 + d];
  } else {
    const float* x = qr + t*128 + h*16;
    val = (__bf16)rope_rotate(x, d - 624, (int)(t & 2047));
  }
  q[i] = val;
}

// same for k; also emits rotated k_rot (fp32) as reference output #3
__global__ __launch_bounds__(256) void assemble_k_kernel(
    const __bf16* __restrict__ kb, const float* __restrict__ kr,
    __bf16* __restrict__ kfull, float* __restrict__ krot_out)
{
  const long i = (long)blockIdx.x * 256 + threadIdx.x;
  const long t = i / 5120;
  const int  c = (int)(i - t*5120);
  const int  h = c / 640, d = c - h*640;
  __bf16 val;
  if (d < 624) {
    val = kb[t*4992 + h*624 + d];
  } else {
    const int j = d - 624;
    const float* x = kr + t*128 + h*16;
    const float o = rope_rotate(x, j, (int)(t & 2047));
    krot_out[t*128 + h*16 + j] = o;
    val = (__bf16)o;
  }
  kfull[i] = val;
}

// -------------------- row softmax: fp32 scores -> bf16 P --------------------
__global__ __launch_bounds__(256) void softmax_rows(const float* __restrict__ S,
                                                    __bf16* __restrict__ P, int cols){
  const long row = blockIdx.x;
  const float* s = S + row * (long)cols;
  __bf16* p = P + row * (long)cols;
  __shared__ float red[8];
  const int tid = threadIdx.x, lane = tid & 31, w = tid >> 5;

  float mx = -3.4e38f;
  for (int i = tid; i < cols; i += 256) mx = fmaxf(mx, s[i]);
#pragma unroll
  for (int o=16; o; o>>=1) mx = fmaxf(mx, __shfl_xor(mx, o, 32));
  if (lane==0) red[w] = mx;
  __syncthreads();
  float m8 = red[0];
#pragma unroll
  for (int i=1;i<8;++i) m8 = fmaxf(m8, red[i]);
  __syncthreads();

  float sum = 0.f;
  for (int i = tid; i < cols; i += 256) sum += __expf(s[i] - m8);
#pragma unroll
  for (int o=16; o; o>>=1) sum += __shfl_xor(sum, o, 32);
  if (lane==0) red[w] = sum;
  __syncthreads();
  float tot = 0.f;
#pragma unroll
  for (int i=0;i<8;++i) tot += red[i];
  const float inv = 1.0f / tot;
  for (int i = tid; i < cols; i += 256) p[i] = (__bf16)(__expf(s[i]-m8)*inv);
}

// ==================== host side ====================
extern "C" void kernel_launch(void* const* d_in, const int* in_sizes, int n_in,
                              void* d_out, int out_size, void* d_ws, size_t ws_size,
                              hipStream_t stream) {
  (void)in_sizes; (void)n_in; (void)out_size; (void)ws_size;
  const long BSN = 4096;                 // B*S tokens
  const long DM = 5120, DKV = 128, DH = 640, NHS = 4992, S = 2048;
  const long SS = S*S;

  const float* h     = (const float*)d_in[0];
  const float* W_dkv = (const float*)d_in[1];  const float* b_dkv = (const float*)d_in[2];
  const float* W_dq  = (const float*)d_in[3];  const float* b_dq  = (const float*)d_in[4];
  const float* W_uk  = (const float*)d_in[5];  const float* b_uk  = (const float*)d_in[6];
  const float* W_uv  = (const float*)d_in[7];  const float* b_uv  = (const float*)d_in[8];
  const float* W_uq  = (const float*)d_in[9];  const float* b_uq  = (const float*)d_in[10];
  const float* W_qr  = (const float*)d_in[11]; const float* b_qr  = (const float*)d_in[12];
  const float* W_kr  = (const float*)d_in[13]; const float* b_kr  = (const float*)d_in[14];
  const float* W_o   = (const float*)d_in[15]; const float* b_o   = (const float*)d_in[16];

  float* out_final = (float*)d_out;                         // [4096,5120]
  float* out_ckv   = out_final + BSN*DM;                    // [4096,128]
  float* out_krot  = out_ckv   + BSN*DKV;                   // [4096,128]

  char* ws = (char*)d_ws;
  size_t cur = 0;
  auto take = [&](size_t bytes)->char* {
    char* p = ws + cur;
    cur += (bytes + 255) & ~(size_t)255;
    return p;
  };
  __bf16* h_bf   = (__bf16*)take(BSN*DM*2);
  __bf16* wt_dkv = (__bf16*)take(DKV*DM*2);
  __bf16* wt_dq  = (__bf16*)take(DKV*DM*2);
  __bf16* wt_uk  = (__bf16*)take(NHS*DKV*2);
  __bf16* wt_uv  = (__bf16*)take(DM*DKV*2);
  __bf16* wt_uq  = (__bf16*)take(NHS*DKV*2);
  __bf16* wt_qr  = (__bf16*)take(DKV*DKV*2);
  __bf16* wt_kr  = (__bf16*)take(DKV*DM*2);
  __bf16* wt_o   = (__bf16*)take(DM*DM*2);
  __bf16* ckv_bf = (__bf16*)take(BSN*DKV*2);
  __bf16* cq_bf  = (__bf16*)take(BSN*DKV*2);
  __bf16* qbase  = (__bf16*)take(BSN*NHS*2);
  __bf16* kbase  = (__bf16*)take(BSN*NHS*2);
  float*  qrot   = (float*) take(BSN*DKV*4);
  float*  krotr  = (float*) take(BSN*DKV*4);
  __bf16* qfull  = (__bf16*)take(BSN*DM*2);
  __bf16* kfull  = (__bf16*)take(BSN*DM*2);
  __bf16* vt     = (__bf16*)take(BSN*DM*2);                 // [b,h,640,2048]
  float*  scores = (float*) take(16*SS*4);
  __bf16* probs  = (__bf16*)take(16*SS*2);
  __bf16* attn_o = (__bf16*)take(BSN*DM*2);                 // [t, h*640+d]

  // 1) h -> bf16
  cvt_f32_to_bf16<<<dim3((unsigned)((BSN*DM)/4/256)), 256, 0, stream>>>(h, h_bf, BSN*DM);

  // 2) weight transpose+convert: W[K][N] -> Wt[N][K] bf16
  auto tr = [&](const float* W, __bf16* Wt, int Kd, int Nd){
    transpose_cvt<<<dim3((Nd+31)/32, (Kd+31)/32), 256, 0, stream>>>(W, Wt, Kd, Nd);
  };
  tr(W_dkv, wt_dkv, 5120, 128);
  tr(W_dq,  wt_dq,  5120, 128);
  tr(W_uk,  wt_uk,  128, 4992);
  tr(W_uv,  wt_uv,  128, 5120);
  tr(W_uq,  wt_uq,  128, 4992);
  tr(W_qr,  wt_qr,  128, 128);
  tr(W_kr,  wt_kr,  5120, 128);
  tr(W_o,   wt_o,   5120, 5120);

  // 3) projection GEMMs
  gemm_bf16_wmma<2><<<dim3(1, 32, 1), 256, 0, stream>>>(      // c_kv: fp32 out + bf16
      h_bf, DM, 0, 0, wt_dkv, DM, 0, 0, b_dkv, 1.0f,
      out_ckv, ckv_bf, DKV, 0, 0, (int)DM);
  gemm_bf16_wmma<1><<<dim3(1, 32, 1), 256, 0, stream>>>(      // c_q
      h_bf, DM, 0, 0, wt_dq, DM, 0, 0, b_dq, 1.0f,
      nullptr, cq_bf, DKV, 0, 0, (int)DM);
  gemm_bf16_wmma<0><<<dim3(1, 32, 1), 256, 0, stream>>>(      // k_rot raw fp32
      h_bf, DM, 0, 0, wt_kr, DM, 0, 0, b_kr, 1.0f,
      krotr, nullptr, DKV, 0, 0, (int)DM);
  gemm_bf16_wmma<0><<<dim3(1, 32, 1), 256, 0, stream>>>(      // q_rot raw fp32
      cq_bf, DKV, 0, 0, wt_qr, DKV, 0, 0, b_qr, 1.0f,
      qrot, nullptr, DKV, 0, 0, (int)DKV);
  gemm_bf16_wmma<1><<<dim3(39, 32, 1), 256, 0, stream>>>(     // k_base
      ckv_bf, DKV, 0, 0, wt_uk, DKV, 0, 0, b_uk, 1.0f,
      nullptr, kbase, NHS, 0, 0, (int)DKV);
  gemm_bf16_wmma<1><<<dim3(39, 32, 1), 256, 0, stream>>>(     // q_base
      cq_bf, DKV, 0, 0, wt_uq, DKV, 0, 0, b_uq, 1.0f,
      nullptr, qbase, NHS, 0, 0, (int)DKV);
  gemm_bf16_wmma<3><<<dim3(40, 32, 1), 256, 0, stream>>>(     // v -> vt[b,h,d,s]
      ckv_bf, DKV, 0, 0, wt_uv, DKV, 0, 0, b_uv, 1.0f,
      nullptr, vt, 0, 0, 0, (int)DKV);

  // 4) rotary + concat
  assemble_q_kernel<<<dim3((unsigned)((BSN*DM)/256)), 256, 0, stream>>>(qbase, qrot, qfull);
  assemble_k_kernel<<<dim3((unsigned)((BSN*DM)/256)), 256, 0, stream>>>(kbase, krotr, kfull, out_krot);

  // 5) scores[b,h] = (Q[b,:,h,:] @ K[b,:,h,:]^T) / sqrt(640)
  const float iscale = 0.03952847075210474f; // 1/sqrt(640)
  gemm_bf16_wmma<0><<<dim3(16, 16, 16), 256, 0, stream>>>(
      qfull, DM, S*DM, DH, kfull, DM, S*DM, DH, nullptr, iscale,
      scores, nullptr, S, 8*SS, SS, (int)DH);

  // 6) softmax rows -> bf16 P
  softmax_rows<<<dim3((unsigned)(16*S)), 256, 0, stream>>>(scores, probs, (int)S);

  // 7) attn_out[t, h*640+d] = P[b,h] @ V[b,h]  (Bt = vt is [d][s] row-major)
  gemm_bf16_wmma<1><<<dim3(5, 16, 16), 256, 0, stream>>>(
      probs, S, 8*SS, SS, vt, S, 8*DH*S, DH*S, nullptr, 1.0f,
      nullptr, attn_o, DM, S*DM, DH, (int)S);

  // 8) final = attn_out @ W_o + b_o  (fp32 to d_out)
  gemm_bf16_wmma<0><<<dim3(40, 32, 1), 256, 0, stream>>>(
      attn_o, DM, 0, 0, wt_o, DM, 0, 0, b_o, 1.0f,
      out_final, nullptr, DM, 0, 0, (int)DM);
}
